// BertAttention_51281909514295
// MI455X (gfx1250) — matrix operand order repaired and checked by reading, not compiled, not run on hardware
//
#include <hip/hip_runtime.h>
#include <hip/hip_bf16.h>
#include <stdint.h>

#define HID   768
#define NH    12
#define HS    64
#define SEQ   512
#define BATCH 32
#define MROWS (BATCH * SEQ)   // 16384

typedef __attribute__((ext_vector_type(16))) __bf16 v16bf;
typedef __attribute__((ext_vector_type(8)))  float  v8f;

static __device__ __forceinline__ unsigned short f2bf_bits(float f) {
  union { float f; unsigned u; } a; a.f = f;
  unsigned r = a.u + 0x7FFFu + ((a.u >> 16) & 1u);   // round-to-nearest-even
  return (unsigned short)(r >> 16);
}

// 16 bf16 lane-fragment overlaid on two 16B vectors: bitcast, no element moves
union VB16 {
  uint4 q[2];
  unsigned u[8];
  unsigned short us[16];
  v16bf v;
};

// fragment from two contiguous 16B bf16 chunks (elements 0-7 / 8-15)
static __device__ __forceinline__ v16bf load_frag_bf(const unsigned short* p0,
                                                     const unsigned short* p1) {
  VB16 t; t.q[0] = *(const uint4*)p0; t.q[1] = *(const uint4*)p1; return t.v;
}

// ---------------------------------------------------------------------------
// Kernel 0: one-time f32 -> bf16 (RNE) conversion, 8 elements per thread.
// Removes all conversion VALU from the GEMM inner loops.
// ---------------------------------------------------------------------------
__global__ void f32_to_bf16_kernel(const float* __restrict__ src,
                                   unsigned short* __restrict__ dst,
                                   int n8) {
  const int idx = blockIdx.x * blockDim.x + threadIdx.x;
  if (idx >= n8) return;
  const float4 a = ((const float4*)src)[idx * 2];
  const float4 b = ((const float4*)src)[idx * 2 + 1];
  uint4 o;
  o.x = ((unsigned)f2bf_bits(a.y) << 16) | f2bf_bits(a.x);
  o.y = ((unsigned)f2bf_bits(a.w) << 16) | f2bf_bits(a.z);
  o.z = ((unsigned)f2bf_bits(b.y) << 16) | f2bf_bits(b.x);
  o.w = ((unsigned)f2bf_bits(b.w) << 16) | f2bf_bits(b.z);
  ((uint4*)dst)[idx] = o;
}

// ---------------------------------------------------------------------------
// Kernel 1: QKV projection, 64x32 register tile per wave, all-bf16 operands.
// y = x @ W^T + b -> bf16 workspace. Per k-step: 2 B-frags held, A-frags
// streamed -> 8 x v_wmma_f32_16x16x32_bf16 per 6 fragment loads (bitcast only).
// mode: 0 = plain (K), 1 = scale by 1/sqrt(64) (Q), 2 = store transposed (V)
// ---------------------------------------------------------------------------
__global__ void qkv_gemm_kernel(const unsigned short* __restrict__ xb,
                                const unsigned short* __restrict__ Wb,
                                const float* __restrict__ bias,
                                unsigned short* __restrict__ out,
                                int mode) {
  const int lane  = threadIdx.x & 31;
  const int wave  = threadIdx.x >> 5;
  const int ntile = blockIdx.x * 4 + wave;   // 0..23 (32-wide N tiles)
  const int mtile = blockIdx.y;              // 0..255 (64-wide M tiles)
  const int m0 = mtile * 64, n0 = ntile * 32;

  const int am  = lane & 15;          // A: row within 16-tile
  const int akb = (lane >> 4) * 16;   // A: K sub-offset per lane-half
  const int bn  = lane & 15;          // B: col within 16-tile
  const int bkb = (lane >> 4) * 16;   // B: K sub-offset per lane-half
  const int hi  = lane >> 4;

  const unsigned short* arow0 = xb + (size_t)(m0 + am) * HID;
  const unsigned short* brow0 = Wb + (size_t)(n0 + bn) * HID;  // W row n == B^T col n
  const unsigned short* brow1 = Wb + (size_t)(n0 + 16 + bn) * HID;

  v8f acc[4][2];
#pragma unroll
  for (int i = 0; i < 4; ++i) { acc[i][0] = (v8f){}; acc[i][1] = (v8f){}; }

  for (int k0 = 0; k0 < HID; k0 += 32) {
    __builtin_prefetch(arow0 + k0 + 64, 0, 1);
    __builtin_prefetch(brow0 + k0 + 64, 0, 1);
    const v16bf B0 = load_frag_bf(brow0 + k0 + bkb, brow0 + k0 + bkb + 8);
    const v16bf B1 = load_frag_bf(brow1 + k0 + bkb, brow1 + k0 + bkb + 8);
#pragma unroll
    for (int i = 0; i < 4; ++i) {
      const unsigned short* ar = arow0 + (size_t)i * 16 * HID + k0;
      const v16bf A = load_frag_bf(ar + akb, ar + akb + 8);
      acc[i][0] = __builtin_amdgcn_wmma_f32_16x16x32_bf16(
          false, A, false, B0, (short)0, acc[i][0], false, false);
      acc[i][1] = __builtin_amdgcn_wmma_f32_16x16x32_bf16(
          false, A, false, B1, (short)0, acc[i][1], false, false);
    }
  }

#pragma unroll
  for (int i = 0; i < 4; ++i) {
#pragma unroll
    for (int j = 0; j < 2; ++j) {
      const int gn = n0 + j * 16 + bn;
      const float bval = bias[gn];
#pragma unroll
      for (int r = 0; r < 8; ++r) {
        int gm = m0 + i * 16 + r + 8 * hi;
        float v = acc[i][j][r] + bval;
        if (mode == 1) v *= 0.125f;                  // fold 1/sqrt(HS) into Q
        if (mode == 2) {                             // V transposed: [b][h][d][s]
          int bb = gm >> 9, ss = gm & 511;
          int h  = gn >> 6, dd = gn & 63;
          out[((size_t)((bb * NH + h) * HS + dd)) * SEQ + ss] = f2bf_bits(v);
        } else {
          out[(size_t)gm * HID + gn] = f2bf_bits(v);
        }
      }
    }
  }
}

// ---------------------------------------------------------------------------
// Kernel 2: flash-style attention. One wave per (batch, head, 16-query tile).
// Online softmax over 32-key blocks; P transposed C-layout -> A-layout via LDS.
// ---------------------------------------------------------------------------
__global__ void attention_kernel(const unsigned short* __restrict__ Qb,
                                 const unsigned short* __restrict__ Kb,
                                 const unsigned short* __restrict__ Vt,
                                 const float* __restrict__ mask,
                                 unsigned short* __restrict__ Ow) {
  __shared__ unsigned short lds_p[16 * 32];   // 16 queries x 32 keys, bf16

  const int lane = threadIdx.x & 31;
  const int qt = blockIdx.x, h = blockIdx.y, b = blockIdx.z;
  const int q0 = qt * 16;

  const int am  = lane & 15;
  const int akb = (lane >> 4) * 16;
  const int bn  = lane & 15;
  const int bkb = (lane >> 4) * 16;
  const int hi  = lane >> 4;

  // Q fragments: K-dim = HS = 64 -> two 16x32 A fragments (pre-scaled by 1/8)
  const unsigned short* qrow = Qb + (size_t)(b * SEQ + q0 + am) * HID + h * HS;
  const v16bf aq0 = load_frag_bf(qrow + akb,      qrow + 8 + akb);
  const v16bf aq1 = load_frag_bf(qrow + 32 + akb, qrow + 40 + akb);

  float run_max[8], run_sum[8];
  v8f o0 = {}, o1 = {}, o2 = {}, o3 = {};
#pragma unroll
  for (int r = 0; r < 8; ++r) { run_max[r] = -1e30f; run_sum[r] = 0.0f; }

  const unsigned short* vbase = Vt + ((size_t)(b * NH + h) * HS) * SEQ;

  for (int kp = 0; kp < SEQ / 32; ++kp) {
    const int key0 = kp * 32;
    v8f s[2];
#pragma unroll
    for (int t = 0; t < 2; ++t) {
      const int kt = key0 + t * 16;
      // B = K^T: per lane bn, contiguous along d within K row
      const unsigned short* krow = Kb + (size_t)(b * SEQ + kt + bn) * HID + h * HS;
      const v16bf b0 = load_frag_bf(krow + bkb,      krow + bkb + 8);
      const v16bf b1 = load_frag_bf(krow + 32 + bkb, krow + 32 + bkb + 8);
      v8f st = {};
      st = __builtin_amdgcn_wmma_f32_16x16x32_bf16(false, aq0, false, b0,
                                                   (short)0, st, false, false);
      st = __builtin_amdgcn_wmma_f32_16x16x32_bf16(false, aq1, false, b1,
                                                   (short)0, st, false, false);
      const float mval = mask[(size_t)b * SEQ + kt + bn];   // (B,1,1,S) broadcast
#pragma unroll
      for (int r = 0; r < 8; ++r) st[r] += mval;
      s[t] = st;
    }

    // online softmax: row m lives in one 16-lane half at fixed VGPR index
    float corr[8];
#pragma unroll
    for (int r = 0; r < 8; ++r) {
      float mx = fmaxf(s[0][r], s[1][r]);
#pragma unroll
      for (int off = 8; off >= 1; off >>= 1) mx = fmaxf(mx, __shfl_xor(mx, off, 32));
      float nm = fmaxf(run_max[r], mx);
      corr[r] = __expf(run_max[r] - nm);
      run_max[r] = nm;
      float p0 = __expf(s[0][r] - nm);
      float p1 = __expf(s[1][r] - nm);
      s[0][r] = p0; s[1][r] = p1;
      float ps = p0 + p1;
#pragma unroll
      for (int off = 8; off >= 1; off >>= 1) ps += __shfl_xor(ps, off, 32);
      run_sum[r] = run_sum[r] * corr[r] + ps;
      o0[r] *= corr[r]; o1[r] *= corr[r]; o2[r] *= corr[r]; o3[r] *= corr[r];
    }

    // P: C-layout regs -> row-major LDS tile -> A-layout fragment
#pragma unroll
    for (int r = 0; r < 8; ++r) {
      int row = r + 8 * hi;
      lds_p[row * 32 + bn]      = f2bf_bits(s[0][r]);
      lds_p[row * 32 + 16 + bn] = f2bf_bits(s[1][r]);
    }
    __syncthreads();
    const v16bf aP = load_frag_bf(&lds_p[am * 32 + akb], &lds_p[am * 32 + 8 + akb]);
    __syncthreads();

    // weighted += P @ V : 4 d-tiles of 16, B from transposed V (contiguous keys)
#pragma unroll
    for (int dt = 0; dt < 4; ++dt) {
      const unsigned short* vrow = vbase + (size_t)(dt * 16 + bn) * SEQ + key0 + bkb;
      const v16bf bv = load_frag_bf(vrow, vrow + 8);
      v8f* op = (dt == 0) ? &o0 : (dt == 1) ? &o1 : (dt == 2) ? &o2 : &o3;
      *op = __builtin_amdgcn_wmma_f32_16x16x32_bf16(false, aP, false, bv,
                                                    (short)0, *op, false, false);
    }
  }

#pragma unroll
  for (int r = 0; r < 8; ++r) {
    float inv = 1.0f / run_sum[r];
    o0[r] *= inv; o1[r] *= inv; o2[r] *= inv; o3[r] *= inv;
  }
#pragma unroll
  for (int r = 0; r < 8; ++r) {
    int gm = b * SEQ + q0 + r + 8 * hi;
    size_t base = (size_t)gm * HID + h * HS + bn;
    Ow[base]      = f2bf_bits(o0[r]);
    Ow[base + 16] = f2bf_bits(o1[r]);
    Ow[base + 32] = f2bf_bits(o2[r]);
    Ow[base + 48] = f2bf_bits(o3[r]);
  }
}

// ---------------------------------------------------------------------------
// Kernel 3: out = weighted @ Wo^T + bo + x (fp32 into d_out), 64x32 reg tile.
// ---------------------------------------------------------------------------
__global__ void out_proj_kernel(const unsigned short* __restrict__ Ww,
                                const unsigned short* __restrict__ Wob,
                                const float* __restrict__ bo,
                                const float* __restrict__ x,
                                float* __restrict__ out) {
  const int lane  = threadIdx.x & 31;
  const int wave  = threadIdx.x >> 5;
  const int ntile = blockIdx.x * 4 + wave;
  const int mtile = blockIdx.y;
  const int m0 = mtile * 64, n0 = ntile * 32;

  const int am  = lane & 15;
  const int akb = (lane >> 4) * 16;
  const int bn  = lane & 15;
  const int bkb = (lane >> 4) * 16;
  const int hi  = lane >> 4;

  const unsigned short* arow0 = Ww + (size_t)(m0 + am) * HID;
  const unsigned short* brow0 = Wob + (size_t)(n0 + bn) * HID;
  const unsigned short* brow1 = Wob + (size_t)(n0 + 16 + bn) * HID;

  v8f acc[4][2];
#pragma unroll
  for (int i = 0; i < 4; ++i) { acc[i][0] = (v8f){}; acc[i][1] = (v8f){}; }

  for (int k0 = 0; k0 < HID; k0 += 32) {
    __builtin_prefetch(arow0 + k0 + 64, 0, 1);
    __builtin_prefetch(brow0 + k0 + 64, 0, 1);
    const v16bf B0 = load_frag_bf(brow0 + k0 + bkb, brow0 + k0 + bkb + 8);
    const v16bf B1 = load_frag_bf(brow1 + k0 + bkb, brow1 + k0 + bkb + 8);
#pragma unroll
    for (int i = 0; i < 4; ++i) {
      const unsigned short* ar = arow0 + (size_t)i * 16 * HID + k0;
      const v16bf A = load_frag_bf(ar + akb, ar + akb + 8);
      acc[i][0] = __builtin_amdgcn_wmma_f32_16x16x32_bf16(
          false, A, false, B0, (short)0, acc[i][0], false, false);
      acc[i][1] = __builtin_amdgcn_wmma_f32_16x16x32_bf16(
          false, A, false, B1, (short)0, acc[i][1], false, false);
    }
  }

#pragma unroll
  for (int i = 0; i < 4; ++i) {
#pragma unroll
    for (int j = 0; j < 2; ++j) {
      const int gn = n0 + j * 16 + bn;
      const float bval = bo[gn];
#pragma unroll
      for (int r = 0; r < 8; ++r) {
        int gm = m0 + i * 16 + r + 8 * hi;
        out[(size_t)gm * HID + gn] = acc[i][j][r] + bval + x[(size_t)gm * HID + gn];
      }
    }
  }
}

// ---------------------------------------------------------------------------
// Kernel 4: in-place LayerNorm over HID per row, then affine (ln_w, ln_b)
// ---------------------------------------------------------------------------
__global__ void layernorm_kernel(float* __restrict__ y,
                                 const float* __restrict__ w,
                                 const float* __restrict__ bias) {
  __shared__ float red[256];
  const int row = blockIdx.x;
  const int t = threadIdx.x;
  float* p = y + (size_t)row * HID;

  float v[3];
  float s = 0.0f, sq = 0.0f;
#pragma unroll
  for (int i = 0; i < 3; ++i) {
    v[i] = p[t + i * 256];
    s += v[i]; sq += v[i] * v[i];
  }
  red[t] = s; __syncthreads();
  for (int off = 128; off > 0; off >>= 1) { if (t < off) red[t] += red[t + off]; __syncthreads(); }
  const float mean = red[0] / (float)HID;
  __syncthreads();
  red[t] = sq; __syncthreads();
  for (int off = 128; off > 0; off >>= 1) { if (t < off) red[t] += red[t + off]; __syncthreads(); }
  const float var = red[0] / (float)HID - mean * mean;
  const float rstd = rsqrtf(var + 1e-12f);
#pragma unroll
  for (int i = 0; i < 3; ++i) {
    int c = t + i * 256;
    p[c] = (v[i] - mean) * rstd * w[c] + bias[c];
  }
}

// ---------------------------------------------------------------------------
extern "C" void kernel_launch(void* const* d_in, const int* in_sizes, int n_in,
                              void* d_out, int out_size, void* d_ws, size_t ws_size,
                              hipStream_t stream) {
  const float* x    = (const float*)d_in[0];
  const float* mask = (const float*)d_in[1];
  const float* Wq = (const float*)d_in[2];
  const float* bq = (const float*)d_in[3];
  const float* Wk = (const float*)d_in[4];
  const float* bk = (const float*)d_in[5];
  const float* Wv = (const float*)d_in[6];
  const float* bv = (const float*)d_in[7];
  const float* Wo = (const float*)d_in[8];
  const float* bo = (const float*)d_in[9];
  const float* lnw = (const float*)d_in[10];
  const float* lnb = (const float*)d_in[11];
  float* out = (float*)d_out;

  // workspace layout (bf16 = unsigned short):
  //   Qb, Kb, Vt, Ww : MROWS*HID each (Q/K row-major, V transposed, weighted)
  //   Xb             : MROWS*HID      (x converted once)
  //   Wqb..Wob       : HID*HID each   (weights converted once)
  const size_t elems = (size_t)MROWS * HID;   // 12.58M
  const size_t welems = (size_t)HID * HID;    // 589824
  unsigned short* Qb  = (unsigned short*)d_ws;
  unsigned short* Kb  = Qb + elems;
  unsigned short* Vt  = Kb + elems;
  unsigned short* Ww  = Vt + elems;
  unsigned short* Xb  = Ww + elems;
  unsigned short* Wqb = Xb + elems;
  unsigned short* Wkb = Wqb + welems;
  unsigned short* Wvb = Wkb + welems;
  unsigned short* Wob = Wvb + welems;

  // one-time f32 -> bf16 conversions (8 elems/thread)
  {
    int n8 = (int)(elems / 8);
    f32_to_bf16_kernel<<<dim3((n8 + 255) / 256), 256, 0, stream>>>(x, Xb, n8);
    int w8 = (int)(welems / 8);
    dim3 gw((w8 + 255) / 256);
    f32_to_bf16_kernel<<<gw, 256, 0, stream>>>(Wq, Wqb, w8);
    f32_to_bf16_kernel<<<gw, 256, 0, stream>>>(Wk, Wkb, w8);
    f32_to_bf16_kernel<<<gw, 256, 0, stream>>>(Wv, Wvb, w8);
    f32_to_bf16_kernel<<<gw, 256, 0, stream>>>(Wo, Wob, w8);
  }

  dim3 g1(6, 256), b1(128);   // 24 N-tiles(32) x 256 M-tiles(64)
  qkv_gemm_kernel<<<g1, b1, 0, stream>>>(Xb, Wqb, bq, Qb, 1);
  qkv_gemm_kernel<<<g1, b1, 0, stream>>>(Xb, Wkb, bk, Kb, 0);
  qkv_gemm_kernel<<<g1, b1, 0, stream>>>(Xb, Wvb, bv, Vt, 2);
  attention_kernel<<<dim3(SEQ / 16, NH, BATCH), 32, 0, stream>>>(Qb, Kb, Vt, mask, Ww);
  out_proj_kernel<<<g1, b1, 0, stream>>>(Ww, Wob, bo, x, out);
  layernorm_kernel<<<dim3(MROWS), 256, 0, stream>>>(out, lnw, lnb);
}
